// ScaledDotProductAttention_69672959475936
// MI455X (gfx1250) — matrix operand order repaired and checked by reading, not compiled, not run on hardware
//
#include <hip/hip_runtime.h>
#include <hip/hip_bf16.h>
#include <stdint.h>

// Problem constants (from reference)
#define NB   4
#define NH   16
#define LQ   2048
#define LK   2048
#define DD   128
#define KT   128                 // K/V rows staged per tile
#define NKT  (LK / KT)           // 16 tiles
#define SCALE 0.08838834764831845f   // 1/sqrt(128)
#define NEGV  (-1e9f)

typedef __attribute__((ext_vector_type(16))) _Float16 v16h;
typedef __attribute__((ext_vector_type(8)))  float    v8f;

union Frag {
    v16h v;
    uint4 q[2];
    unsigned short h[16];
};

__device__ __forceinline__ unsigned short f2h(float f) {
    union { _Float16 h; unsigned short u; } c;
    c.h = (_Float16)f;           // v_cvt_f16_f32
    return c.u;
}

// ---- LDS layout (dynamic shared) -------------------------------------------
// Sbuf  : float [16][2048]          131072 B   scores -> exp values
// Qs    : f16   [16][128]             4096 B
// Kt    : f16   [2][128][128]        65536 B   row-major (k-row, d)
// Vt    : f16   [2][128][128]        65536 B   TRANSPOSED (d, k-row)
// red   : float [16][16]              1024 B
// rowmax: float [16]                    64 B
// rowinv: float [16]                    64 B
#define OFF_S   0
#define OFF_Q   (131072)
#define OFF_K   (OFF_Q + 4096)
#define OFF_V   (OFF_K + 65536)
#define OFF_R   (OFF_V + 65536)
#define OFF_RM  (OFF_R + 1024)
#define OFF_RI  (OFF_RM + 64)
#define SMEM_TOTAL (OFF_RI + 64)

extern __shared__ __align__(16) char smem[];

__global__ __launch_bounds__(256)
void sdpa_fwd_kernel(const float* __restrict__ Qg_, const float* __restrict__ Kg_,
                     const float* __restrict__ Vg_, const unsigned char* __restrict__ maskP,
                     float* __restrict__ ctx, float* __restrict__ attn)
{
    float*          Sbuf   = (float*)(smem + OFF_S);
    unsigned short* Qs     = (unsigned short*)(smem + OFF_Q);
    unsigned short* Ktile  = (unsigned short*)(smem + OFF_K);
    unsigned short* Vtile  = (unsigned short*)(smem + OFF_V);
    float*          red    = (float*)(smem + OFF_R);
    float*          rowmax = (float*)(smem + OFF_RM);
    float*          rowinv = (float*)(smem + OFF_RI);

    const int tid  = threadIdx.x;
    const int lane = tid & 31;
    const int w    = tid >> 5;          // wave id 0..7

    const int bh    = blockIdx.y;       // fused (b,h) 0..63
    const int qblk  = blockIdx.x;       // 0..127
    const int qbase = qblk * 16;

    const float* Qg = Qg_ + ((size_t)bh * LQ + qbase) * DD;
    const float* Kg = Kg_ + (size_t)bh * LK * DD;
    const float* Vg = Vg_ + (size_t)bh * LK * DD;
    const unsigned char* Mg = maskP + ((size_t)bh * LQ + qbase) * LK;
    float* Ag = attn + ((size_t)bh * LQ + qbase) * LK;
    float* Cg = ctx  + ((size_t)bh * LQ + qbase) * DD;

    // ---- stage Q strip: 16x128 f32 -> f16 ----------------------------------
    for (int i = 0; i < 2; ++i) {
        int idx = tid + i * 256;            // 512 float4 total
        int r  = idx >> 5;                  // 32 float4 per row
        int c4 = idx & 31;
        float4 f = ((const float4*)Qg)[(size_t)r * 32 + c4];
        uint2 p;
        p.x = (uint32_t)f2h(f.x) | ((uint32_t)f2h(f.y) << 16);
        p.y = (uint32_t)f2h(f.z) | ((uint32_t)f2h(f.w) << 16);
        *(uint2*)&Qs[r * DD + c4 * 4] = p;
    }
    __syncthreads();

    // ---- per-wave Q A-fragments (16-bit A layout, ISA 7.12.2) --------------
    // lane<16: row=lane,    K = {d0+0..7,  d0+16..23}
    // lane>=16: row=lane-16, K = {d0+8..15, d0+24..31}
    Frag qa[4];
    {
        const int m    = lane & 15;
        const int koff = (lane >> 4) * 8;
        for (int c = 0; c < 4; ++c) {
            const int d0 = c * 32;
            qa[c].q[0] = *(const uint4*)&Qs[m * DD + d0 + koff];
            qa[c].q[1] = *(const uint4*)&Qs[m * DD + d0 + 16 + koff];
        }
    }

    // ======================= Pass 1: S = mask(scale * Q K^T) ================
    const int n    = lane & 15;
    const int bsel = (lane >> 4) * 8;       // B-fragment K-half select
    for (int kt = 0; kt < NKT; ++kt) {
        unsigned short* Kb = Ktile + (kt & 1) * (KT * DD);
        const float* Ksrc = Kg + (size_t)kt * KT * DD;
        if (kt + 1 < NKT) {                 // pull next tile toward L2
            const float* nxt = Ksrc + (size_t)KT * DD;
            for (int p = 0; p < 4; ++p)
                __builtin_prefetch(nxt + p * 4096 + tid * 16, 0, 0);
        }
        for (int i = 0; i < 16; ++i) {      // 128x128 f32 -> f16, coalesced
            int idx = tid + i * 256;        // 4096 float4
            int r  = idx >> 5;
            int c4 = idx & 31;
            float4 f = ((const float4*)Ksrc)[(size_t)r * 32 + c4];
            uint2 p;
            p.x = (uint32_t)f2h(f.x) | ((uint32_t)f2h(f.y) << 16);
            p.y = (uint32_t)f2h(f.z) | ((uint32_t)f2h(f.w) << 16);
            *(uint2*)&Kb[r * DD + c4 * 4] = p;
        }
        __syncthreads();

        // wave w computes 16x16 S-tile at columns kt*128 + w*16
        v8f acc = {};
        for (int c = 0; c < 4; ++c) {
            const int d0 = c * 32;
            Frag kb;
            const unsigned short* kr = &Kb[(w * 16 + n) * DD + d0 + bsel];
            kb.q[0] = *(const uint4*)kr;
            kb.q[1] = *(const uint4*)(kr + 16);
            acc = __builtin_amdgcn_wmma_f32_16x16x32_f16(
                false, qa[c].v, false, kb.v, (short)0, acc, false, false);
        }
        const int col = kt * KT + w * 16 + n;
        for (int r = 0; r < 8; ++r) {
            const int mrow = r + ((lane >> 4) << 3);
            unsigned char mk = Mg[(size_t)mrow * LK + col];
            Sbuf[mrow * LK + col] = mk ? NEGV : acc[r] * SCALE;
        }
    }
    __syncthreads();

    // ======================= softmax over the 16x2048 strip =================
    {
        const int row = tid >> 4;           // 16 threads per row
        const int seg = tid & 15;           // 128 elements per segment
        float* Sr = Sbuf + row * LK + seg * 128;
        float m = -3.4e38f;
        for (int c = 0; c < 128; ++c) m = fmaxf(m, Sr[c]);
        red[row * 16 + seg] = m;
        __syncthreads();
        if (tid < 16) {
            float mm = red[tid * 16];
            for (int s = 1; s < 16; ++s) mm = fmaxf(mm, red[tid * 16 + s]);
            rowmax[tid] = mm;
        }
        __syncthreads();
        const float rm = rowmax[row];
        float sum = 0.f;
        for (int c = 0; c < 128; ++c) {
            float e = __expf(Sr[c] - rm);
            Sr[c] = e;                      // keep unnormalized exp for P.V
            sum += e;
        }
        red[row * 16 + seg] = sum;
        __syncthreads();
        if (tid < 16) {
            float ss = 0.f;
            for (int s = 0; s < 16; ++s) ss += red[tid * 16 + s];
            rowinv[tid] = 1.0f / ss;
        }
        __syncthreads();
    }

    // ---- write normalized attention strip (coalesced float4) ---------------
    for (int i = 0; i < 32; ++i) {
        int idx = tid + i * 256;            // 8192 float4 total
        int row = idx >> 9;                 // 512 float4 per row
        int c4  = idx & 511;
        const float rinv = rowinv[row];
        const float* s = Sbuf + row * LK + c4 * 4;
        float4 o = { s[0] * rinv, s[1] * rinv, s[2] * rinv, s[3] * rinv };
        ((float4*)Ag)[(size_t)row * 512 + c4] = o;
    }

    // ======================= Pass 2: O = P V ================================
    v8f oacc = {};
    const int d0w = w * 16;                 // wave-owned 16-wide d tile
    for (int kt = 0; kt < NKT; ++kt) {
        unsigned short* Vb = Vtile + (kt & 1) * (DD * KT);
        const float* Vsrc = Vg + (size_t)kt * KT * DD;
        if (kt + 1 < NKT) {
            const float* nxt = Vsrc + (size_t)KT * DD;
            for (int p = 0; p < 4; ++p)
                __builtin_prefetch(nxt + p * 4096 + tid * 16, 0, 0);
        }
        for (int i = 0; i < 16; ++i) {      // load row-major, store transposed
            int idx = tid + i * 256;
            int r  = idx >> 5;
            int c4 = idx & 31;
            float4 f = ((const float4*)Vsrc)[(size_t)r * 32 + c4];
            const int c = c4 * 4;
            Vb[(c + 0) * KT + r] = f2h(f.x);
            Vb[(c + 1) * KT + r] = f2h(f.y);
            Vb[(c + 2) * KT + r] = f2h(f.z);
            Vb[(c + 3) * KT + r] = f2h(f.w);
        }
        __syncthreads();

        const int m    = lane & 15;
        const int koff = (lane >> 4) * 8;
        for (int ks = 0; ks < 4; ++ks) {    // K in chunks of 32
            const int kb = kt * KT + ks * 32;
            Frag pa;                         // A = exp(S) strip, f32 -> f16
            const float* Sr = Sbuf + m * LK + kb;
            for (int i = 0; i < 8; ++i) {
                pa.h[i]     = f2h(Sr[koff + i]);
                pa.h[8 + i] = f2h(Sr[16 + koff + i]);
            }
            Frag vb;                         // B = V^T strip, contiguous per lane
            const int kl = ks * 32 + bsel;
            const unsigned short* v0 = &Vb[(d0w + m) * KT + kl];
            vb.q[0] = *(const uint4*)v0;
            vb.q[1] = *(const uint4*)(v0 + 16);
            oacc = __builtin_amdgcn_wmma_f32_16x16x32_f16(
                false, pa.v, false, vb.v, (short)0, oacc, false, false);
        }
    }

    // ---- scale by 1/rowsum and store context -------------------------------
    for (int r = 0; r < 8; ++r) {
        const int mrow = r + ((lane >> 4) << 3);
        Cg[(size_t)mrow * DD + d0w + (lane & 15)] = oacc[r] * rowinv[mrow];
    }
}

extern "C" void kernel_launch(void* const* d_in, const int* in_sizes, int n_in,
                              void* d_out, int out_size, void* d_ws, size_t ws_size,
                              hipStream_t stream) {
    (void)in_sizes; (void)n_in; (void)d_ws; (void)ws_size; (void)out_size;
    const float* Q = (const float*)d_in[0];
    const float* K = (const float*)d_in[1];
    const float* V = (const float*)d_in[2];
    const unsigned char* Msk = (const unsigned char*)d_in[3];   // jax bool: 1B/elem

    float* ctx  = (float*)d_out;                                 // [B,H,LQ,D]
    float* attn = ctx + (size_t)NB * NH * LQ * DD;               // [B,H,LQ,LK]

    // Allow the large dynamic LDS allocation (harmless if already allowed).
    (void)hipFuncSetAttribute((const void*)sdpa_fwd_kernel,
                              hipFuncAttributeMaxDynamicSharedMemorySize,
                              SMEM_TOTAL);

    dim3 grid(LQ / 16, NB * NH);
    dim3 block(256);
    sdpa_fwd_kernel<<<grid, block, SMEM_TOTAL, stream>>>(Q, K, V, Msk, ctx, attn);
}
